// Caser_57827439673482
// MI455X (gfx1250) — compile-verified, matrix operand order checked by reading, not run on hardware
//
#include <hip/hip_runtime.h>

typedef __attribute__((ext_vector_type(16))) _Float16 v16h;
typedef __attribute__((ext_vector_type(8)))  _Float16 v8h;
typedef __attribute__((ext_vector_type(8)))  float    v8f;

#define L_SEQ 31
#define B_SZ  512
#define D_SZ  256
#define H_SZ  512
#define V_SZ  50000
#define NH_SZ 64
#define NV_SZ 64
#define K1    18368      // NV*D + NH*L
#define LD_WH 7936       // L*D (row stride of Wh[i-1][h])

// ---------------- WMMA fragment helpers (wave32, 16x16x32 f16) -------------
// 16-bit A/B layout: lanes 0-15 hold row (lane&15) K=0..7 & 16..23;
// lanes 16-31 hold K=8..15 & 24..31.  Caller passes p = rowBase + k0 + sel*8.
static __device__ __forceinline__ v16h frag_f16(const _Float16* p) {
  union { v16h v; v8h h[2]; } u;
  u.h[0] = *(const v8h*)(p);
  u.h[1] = *(const v8h*)(p + 16);
  return u.v;
}

static __device__ __forceinline__ v16h frag_f32(const float* p) {
  union { v16h v; _Float16 e[16]; } u;
#pragma unroll
  for (int j = 0; j < 8; ++j) {
    u.e[j]     = (_Float16)p[j];
    u.e[8 + j] = (_Float16)p[16 + j];
  }
  return u.v;
}

// 4 M-tiles x 1 N-tile per wave: one B fragment feeds 4 WMMAs.
// A (MxK) row-major; B stored transposed (NxK) row-major.
template <bool AF32, bool BF32>
static __device__ __forceinline__ void gemm_tile4(const void* A, int lda,
                                                  const void* B, int ldb,
                                                  int mbase, int nbase, int K,
                                                  int lane, v8f* acc) {
  const int sel = lane >> 4;
  const int r   = lane & 15;
  const size_t a0 = (size_t)(mbase + r) * lda + sel * 8;
  const size_t as = (size_t)16 * lda;
  const size_t b0 = (size_t)(nbase + r) * ldb + sel * 8;
  for (int k0 = 0; k0 < K; k0 += 32) {
    v16h b;
    if constexpr (BF32) {
      b = frag_f32((const float*)B + b0 + k0);
      __builtin_prefetch((const float*)B + b0 + k0 + 512, 0, 0);
    } else {
      b = frag_f16((const _Float16*)B + b0 + k0);
      __builtin_prefetch((const _Float16*)B + b0 + k0 + 1024, 0, 0);
    }
#pragma unroll
    for (int t = 0; t < 4; ++t) {
      v16h a;
      if constexpr (AF32) a = frag_f32((const float*)A + a0 + t * as + k0);
      else                a = frag_f16((const _Float16*)A + a0 + t * as + k0);
      acc[t] = __builtin_amdgcn_wmma_f32_16x16x32_f16(false, a, false, b,
                                                      (short)0, acc[t], false, false);
    }
  }
}

// ---------------------------- small kernels --------------------------------
__global__ void cvt_kernel(const float* __restrict__ src,
                           _Float16* __restrict__ dst, int n) {
  int i = blockIdx.x * blockDim.x + threadIdx.x;
  if (i < n) dst[i] = (_Float16)src[i];
}

// out_v[b, c*256+d] = bv[c] + sum_l item[b,l,d] * Wv[c,l]
__global__ void outv_kernel(const float* __restrict__ item,
                            const float* __restrict__ Wv,
                            const float* __restrict__ bv,
                            float* __restrict__ out) {
  __shared__ float wv[NV_SZ * L_SEQ];
  for (int i = threadIdx.x; i < NV_SZ * L_SEQ; i += blockDim.x) wv[i] = Wv[i];
  __syncthreads();
  int idx = blockIdx.x * blockDim.x + threadIdx.x;      // over B*D
  int b = idx >> 8, d = idx & 255;
  float it[L_SEQ];
#pragma unroll
  for (int l = 0; l < L_SEQ; ++l) it[l] = item[((size_t)b * L_SEQ + l) * D_SZ + d];
  for (int c = 0; c < NV_SZ; ++c) {
    float acc = bv[c];
#pragma unroll
    for (int l = 0; l < L_SEQ; ++l) acc += it[l] * wv[c * L_SEQ + l];
    out[(size_t)b * K1 + c * D_SZ + d] = acc;
  }
}

// user half of x: x[b, 256+d] = user_emb_table[user[b], d]
__global__ void user_kernel(const int* __restrict__ user,
                            const float* __restrict__ tbl,
                            _Float16* __restrict__ x) {
  int idx = blockIdx.x * blockDim.x + threadIdx.x;      // B*D
  int b = idx >> 8, d = idx & 255;
  x[(size_t)b * 512 + 256 + d] = (_Float16)tbl[(size_t)user[b] * D_SZ + d];
}

// ------------------------- horizontal conv (WMMA) --------------------------
// grid = (32, 31): blockIdx.x = batch tile, blockIdx.y = i-1. 128 threads.
// Wave w owns a fixed 16x16 (batch x channel) tile and loops over positions p,
// max-pooling relu(conv+bias) in registers -> single store, no atomics.
__global__ void conv_kernel(const _Float16* __restrict__ itemF16,
                            const _Float16* __restrict__ WhF16,
                            const float* __restrict__ bh,
                            float* __restrict__ out) {
  const int i = blockIdx.y + 1;                         // conv width 1..31
  const int btile = blockIdx.x;                         // 0..31
  const int lane = threadIdx.x & 31;
  const int wave = threadIdx.x >> 5;                    // channel tile 0..3
  const int sel = lane >> 4;
  const int r = lane & 15;
  const int b = btile * 16 + r;
  const int h = wave * 16 + r;
  const _Float16* aBase = itemF16 + (size_t)b * L_SEQ * D_SZ + sel * 8;
  // Wh[i-1, h, :i, :] is a contiguous prefix of a 7936-half row
  const _Float16* bRow  = WhF16 + ((size_t)(i - 1) * NH_SZ + h) * LD_WH + sel * 8;
  const int K = i * D_SZ;
  const float bias = bh[(i - 1) * NH_SZ + h];
  v8f vmax = {};                                        // relu -> max >= 0
  for (int p = 0; p <= L_SEQ - i; ++p) {
    const _Float16* aRow = aBase + (size_t)p * D_SZ;    // contiguous patch
    v8f acc = {};
    for (int k0 = 0; k0 < K; k0 += 32) {
      v16h a = frag_f16(aRow + k0);
      v16h bb = frag_f16(bRow + k0);
      acc = __builtin_amdgcn_wmma_f32_16x16x32_f16(false, a, false, bb,
                                                   (short)0, acc, false, false);
    }
#pragma unroll
    for (int j = 0; j < 8; ++j) {
      float v = acc[j] + bias;
      v = v > 0.f ? v : 0.f;
      vmax[j] = vmax[j] > v ? vmax[j] : v;
    }
  }
#pragma unroll
  for (int j = 0; j < 8; ++j) {
    int bj = btile * 16 + j + sel * 8;
    out[(size_t)bj * K1 + NV_SZ * D_SZ + (i - 1) * NH_SZ + h] = vmax[j];
  }
}

// ------------------------------ GEMM kernels -------------------------------
// z = relu(out @ W1^T + b1) -> x[:, :256]  (M=512, N=256, K=18368)
__global__ void w1_kernel(const float* __restrict__ out,
                          const float* __restrict__ W1,
                          const float* __restrict__ b1,
                          _Float16* __restrict__ x) {
  int nt = blockIdx.x >> 1;                              // 16 n-tiles, grid 32
  int mg = (blockIdx.x & 1) * 4 + (threadIdx.x >> 5);    // 8 m-groups of 64 rows
  int lane = threadIdx.x & 31, sel = lane >> 4;
  v8f acc[4] = {};
  gemm_tile4<true, true>(out, K1, W1, K1, mg * 64, nt * 16, K1, lane, acc);
  int n = nt * 16 + (lane & 15);
  float bias = b1[n];
#pragma unroll
  for (int t = 0; t < 4; ++t)
#pragma unroll
    for (int j = 0; j < 8; ++j) {
      int m = mg * 64 + t * 16 + j + sel * 8;
      float v = acc[t][j] + bias;
      v = v > 0.f ? v : 0.f;
      x[(size_t)m * 512 + n] = (_Float16)v;
    }
}

// seq_output = relu(x @ W2^T + b2)  (M=512, N=512, K=512); f32 out + f16 copy
__global__ void w2_kernel(const _Float16* __restrict__ x,
                          const float* __restrict__ W2,
                          const float* __restrict__ b2,
                          float* __restrict__ seqOut,
                          _Float16* __restrict__ seqF16) {
  int nt = blockIdx.x >> 1;                              // 32 n-tiles, grid 64
  int mg = (blockIdx.x & 1) * 4 + (threadIdx.x >> 5);
  int lane = threadIdx.x & 31, sel = lane >> 4;
  v8f acc[4] = {};
  gemm_tile4<false, true>(x, 512, W2, 512, mg * 64, nt * 16, 512, lane, acc);
  int n = nt * 16 + (lane & 15);
  float bias = b2[n];
#pragma unroll
  for (int t = 0; t < 4; ++t)
#pragma unroll
    for (int j = 0; j < 8; ++j) {
      int m = mg * 64 + t * 16 + j + sel * 8;
      float v = acc[t][j] + bias;
      v = v > 0.f ? v : 0.f;
      seqOut[(size_t)m * H_SZ + n] = v;
      seqF16[(size_t)m * H_SZ + n] = (_Float16)v;
    }
}

// logit = seq_output @ Wp^T + bp  (M=512, N=50000, K=512)
__global__ void logit_kernel(const _Float16* __restrict__ seqF16,
                             const float* __restrict__ Wp,
                             const float* __restrict__ bp,
                             float* __restrict__ logit) {
  int nt = blockIdx.x >> 1;                              // 3125 n-tiles, grid 6250
  int mg = (blockIdx.x & 1) * 4 + (threadIdx.x >> 5);
  int lane = threadIdx.x & 31, sel = lane >> 4;
  v8f acc[4] = {};
  gemm_tile4<false, true>(seqF16, 512, Wp, 512, mg * 64, nt * 16, 512, lane, acc);
  int n = nt * 16 + (lane & 15);
  float bias = bp[n];
#pragma unroll
  for (int t = 0; t < 4; ++t)
#pragma unroll
    for (int j = 0; j < 8; ++j) {
      int m = mg * 64 + t * 16 + j + sel * 8;
      logit[(size_t)m * V_SZ + n] = acc[t][j] + bias;
    }
}

// hidden_seq = item_seq @ Wsc^T + bsc  (M=B*L=15872, N=512, K=256)
__global__ void hidden_kernel(const _Float16* __restrict__ itemF16,
                              const float* __restrict__ Wsc,
                              const float* __restrict__ bsc,
                              float* __restrict__ hid) {
  int nt = blockIdx.x / 62;                              // 32 n-tiles, grid 1984
  int g  = blockIdx.x % 62;
  int mg = g * 4 + (threadIdx.x >> 5);                   // 248 m-groups of 64 rows
  int lane = threadIdx.x & 31, sel = lane >> 4;
  v8f acc[4] = {};
  gemm_tile4<false, true>(itemF16, 256, Wsc, 256, mg * 64, nt * 16, 256, lane, acc);
  int n = nt * 16 + (lane & 15);
  float bias = bsc[n];
#pragma unroll
  for (int t = 0; t < 4; ++t)
#pragma unroll
    for (int j = 0; j < 8; ++j) {
      int m = mg * 64 + t * 16 + j + sel * 8;
      hid[(size_t)m * H_SZ + n] = acc[t][j] + bias;
    }
}

// ------------------------------- launcher ----------------------------------
extern "C" void kernel_launch(void* const* d_in, const int* in_sizes, int n_in,
                              void* d_out, int out_size, void* d_ws, size_t ws_size,
                              hipStream_t stream) {
  const int*   user      = (const int*)  d_in[0];
  const float* item_seq  = (const float*)d_in[1];
  const float* user_tbl  = (const float*)d_in[2];
  const float* Wv        = (const float*)d_in[3];
  const float* bv        = (const float*)d_in[4];
  const float* Wh        = (const float*)d_in[5];
  const float* bh        = (const float*)d_in[6];
  const float* W1        = (const float*)d_in[7];
  const float* b1        = (const float*)d_in[8];
  const float* W2        = (const float*)d_in[9];
  const float* b2        = (const float*)d_in[10];
  const float* Wp        = (const float*)d_in[11];
  const float* bp        = (const float*)d_in[12];
  const float* Wsc       = (const float*)d_in[13];
  const float* bsc       = (const float*)d_in[14];

  // outputs: logit (512x50000) | hidden_seq (512x31x512) | seq_output (512x512)
  float* logit = (float*)d_out;
  float* hid   = logit + (size_t)B_SZ * V_SZ;
  float* seqO  = hid + (size_t)B_SZ * L_SEQ * H_SZ;

  // workspace layout (~78 MB)
  char* ws = (char*)d_ws;
  _Float16* itemF16 = (_Float16*)ws;                 ws += (size_t)B_SZ * L_SEQ * D_SZ * 2;
  _Float16* whF16   = (_Float16*)ws;                 ws += (size_t)L_SEQ * NH_SZ * L_SEQ * D_SZ * 2;
  float*    outbuf  = (float*)ws;                    ws += (size_t)B_SZ * K1 * 4;
  _Float16* xbuf    = (_Float16*)ws;                 ws += (size_t)B_SZ * 512 * 2;
  _Float16* seqF16  = (_Float16*)ws;                 ws += (size_t)B_SZ * 512 * 2;

  const int nItem = B_SZ * L_SEQ * D_SZ;             // 4,063,232
  const int nWh   = L_SEQ * NH_SZ * L_SEQ * D_SZ;    // 15,745,024

  cvt_kernel<<<(nItem + 255) / 256, 256, 0, stream>>>(item_seq, itemF16, nItem);
  cvt_kernel<<<(nWh + 255) / 256, 256, 0, stream>>>(Wh, whF16, nWh);
  outv_kernel<<<(B_SZ * D_SZ) / 256, 256, 0, stream>>>(item_seq, Wv, bv, outbuf);
  conv_kernel<<<dim3(32, 31), 128, 0, stream>>>(itemF16, whF16, bh, outbuf);
  w1_kernel<<<32, 128, 0, stream>>>(outbuf, W1, b1, xbuf);
  user_kernel<<<(B_SZ * D_SZ) / 256, 256, 0, stream>>>(user, user_tbl, xbuf);
  w2_kernel<<<64, 128, 0, stream>>>(xbuf, W2, b2, seqO, seqF16);
  logit_kernel<<<3125 * 2, 128, 0, stream>>>(seqF16, Wp, bp, logit);
  hidden_kernel<<<32 * 62, 128, 0, stream>>>(itemF16, Wsc, bsc, hid);
}
